// CrossScan_SS2D_23699629539952
// MI455X (gfx1250) — matrix operand order repaired and checked by reading, not compiled, not run on hardware
//
#include <hip/hip_runtime.h>
#include <stdint.h>

// ---------------- problem constants ----------------
#define D_MODEL   192
#define D_STATE   64
#define D_CONV    4
#define HEADDIM   64
#define D_INNER   384          // EXPAND * D_MODEL
#define NHEADS    6            // D_INNER / HEADDIM
#define D_IN_PROJ 902          // 2*D_INNER + 2*D_STATE + NHEADS
#define CONV_DIM  512          // D_INNER + 2*D_STATE
#define EPS       1e-5f

#define BB  4                  // batch
#define HH  32
#define WW  32
#define LL  1024               // H*W
#define NSEQ 16                // 4 directions * batch
#define MROWS (NSEQ*LL)        // 16384 GEMM rows
#define LDZ   912              // padded in_proj row stride (57 tiles * 16)

// WMMA vector types (CDNA5 wave32: 16x16x32 bf16, f32 accum)
typedef __attribute__((ext_vector_type(16))) __bf16 v16bf;
typedef __attribute__((ext_vector_type(8)))  float  v8f;

#define WMMA_BF16(A, B, C) \
  __builtin_amdgcn_wmma_f32_16x16x32_bf16(false, (A), false, (B), (short)0, (C), false, false)

__device__ __forceinline__ uint32_t pack_bf16x2(float a, float b) {
  uint32_t ua = __float_as_uint(a);
  uint32_t ub = __float_as_uint(b);
  ua = (ua + 0x7FFFu + ((ua >> 16) & 1u)) >> 16;   // round-to-nearest-even
  ub = (ub + 0x7FFFu + ((ub >> 16) & 1u)) >> 16;
  return (ua & 0xFFFFu) | (ub << 16);
}

__device__ __forceinline__ float siluf(float x) { return x / (1.f + expf(-x)); }

// =====================================================================
// 1) Gather 4 directional sequences from x[B,32,32,192] directly into the
//    packed-bf16 WMMA A-operand layout (ISA 7.12.2, 16-bit A 16x32):
//    dword index = ((kblk*Mtiles + mtile)*32 + lane)*8 + v
//    lane<16 -> M=lane&15, lane>=16 -> K offset +8
//    v in 0..3: K = 2v + 8*hi ; v in 4..7: K = 16 + 2(v-4) + 8*hi
// =====================================================================
__global__ void gather_pack_A1(const float* __restrict__ x, uint32_t* __restrict__ dst) {
  int idx = blockIdx.x * blockDim.x + threadIdx.x;          // 6*1024*256 total
  int v    = idx & 7;
  int lane = (idx >> 3) & 31;
  int mt   = (idx >> 8) & 1023;
  int kb   = idx >> 18;                                     // 0..5
  int hi = lane >> 4, g = v >> 2, vv = v & 3;
  int m  = mt * 16 + (lane & 15);                           // 0..16383
  int c  = kb * 32 + 16 * g + 2 * vv + 8 * hi;              // channel 0..190
  int s  = m >> 10, l = m & 1023;
  int dir = s >> 2, b = s & 3;
  int i = l >> 5, j = l & 31, h, w;
  switch (dir) {
    case 0:  h = i;      w = j;      break;   // row-major scan
    case 1:  h = i;      w = 31 - j; break;   // row-major, W flipped
    case 2:  h = j;      w = i;      break;   // column-major scan
    default: h = 31 - j; w = i;      break;   // column-major, H flipped
  }
  const float* xr = x + (((size_t)(b * 32 + h)) * 32 + w) * D_MODEL;
  dst[idx] = pack_bf16x2(xr[c], xr[c + 1]);
}

// Pack f32 activations [M][K] row-major -> bf16 WMMA A layout
__global__ void pack_A(const float* __restrict__ src, uint32_t* __restrict__ dst, int K) {
  int idx = blockIdx.x * blockDim.x + threadIdx.x;
  int v    = idx & 7;
  int lane = (idx >> 3) & 31;
  int mt   = (idx >> 8) & 1023;
  int kb   = idx >> 18;
  int hi = lane >> 4, g = v >> 2, vv = v & 3;
  int m  = mt * 16 + (lane & 15);
  int k0 = kb * 32 + 16 * g + 2 * vv + 8 * hi;
  const float* sr = src + (size_t)m * K + k0;
  dst[idx] = pack_bf16x2(sr[0], sr[1]);
}

// Pack f32 weights [K][N] row-major -> bf16 WMMA B layout (16-bit B 32x16):
// lane<16 -> N=lane&15, K=16(v>>2)+2(v&3),+1 ; lane>=16 -> K offset +8.
// NtilesPad >= ceil(N/16): columns n>=N are zero-filled (padding tiles
// contribute exact zeros so the GEMM needs no bounds branches).
__global__ void pack_B(const float* __restrict__ src, uint32_t* __restrict__ dst,
                       int K, int N, int NtilesPad) {
  int idx = blockIdx.x * blockDim.x + threadIdx.x;
  int total = (K / 32) * NtilesPad * 256;
  if (idx >= total) return;
  int v    = idx & 7;
  int lane = (idx >> 3) & 31;
  int nt   = (idx >> 8) % NtilesPad;
  int kb   = (idx >> 8) / NtilesPad;
  int n  = nt * 16 + (lane & 15);
  int k0 = kb * 32 + 16 * (v >> 2) + 2 * (v & 3) + 8 * (lane >> 4);
  float f0 = 0.f, f1 = 0.f;
  if (n < N) {
    f0 = src[(size_t)k0 * N + n];
    f1 = src[(size_t)(k0 + 1) * N + n];
  }
  dst[idx] = pack_bf16x2(f0, f1);
}

// =====================================================================
// 2) bf16 WMMA GEMM, f32 accumulate. One wave -> 32(M) x 64(N) tile:
//    branchless inner loop (EXEC stays all-ones around every WMMA),
//    per k-step: 4x b128 A loads + 8x b128 B loads + 8 WMMAs.
//    Grids are exact multiples (no early-out guard).
// =====================================================================
__global__ void __launch_bounds__(256)
gemm_bf16_wmma(const uint32_t* __restrict__ Apack, const uint32_t* __restrict__ Bpack,
               float* __restrict__ C, int Mtiles, int NtilesPad, int Ngroups,
               int Kblks, int ldc, int ncols) {
  int wid  = blockIdx.x * (blockDim.x >> 5) + (threadIdx.x >> 5);
  int lane = threadIdx.x & 31;
  int mpair = wid / Ngroups;                 // 0 .. Mtiles/2 - 1
  int ngrp  = wid % Ngroups;
  int mt0 = mpair * 2, mt1 = mt0 + 1;

  const uint32_t* a0p = Apack + (((size_t)mt0) * 32 + lane) * 8;
  const uint32_t* a1p = Apack + (((size_t)mt1) * 32 + lane) * 8;
  const uint32_t* bp  = Bpack + (((size_t)(ngrp * 4)) * 32 + lane) * 8;
  const size_t astep = (size_t)Mtiles * 256;
  const size_t bstep = (size_t)NtilesPad * 256;

  v8f c00 = {}, c01 = {}, c02 = {}, c03 = {};
  v8f c10 = {}, c11 = {}, c12 = {}, c13 = {};
  for (int kb = 0; kb < Kblks; ++kb) {
    v16bf a0 = *(const v16bf*)a0p;
    v16bf a1 = *(const v16bf*)a1p;
    v16bf b0 = *(const v16bf*)(bp);
    v16bf b1 = *(const v16bf*)(bp + 256);
    v16bf b2 = *(const v16bf*)(bp + 512);
    v16bf b3 = *(const v16bf*)(bp + 768);
    c00 = WMMA_BF16(a0, b0, c00);
    c10 = WMMA_BF16(a1, b0, c10);
    c01 = WMMA_BF16(a0, b1, c01);
    c11 = WMMA_BF16(a1, b1, c11);
    c02 = WMMA_BF16(a0, b2, c02);
    c12 = WMMA_BF16(a1, b2, c12);
    c03 = WMMA_BF16(a0, b3, c03);
    c13 = WMMA_BF16(a1, b3, c13);
    a0p += astep; a1p += astep; bp += bstep;
  }
  // C/D layout: VGPR r -> row = mtile*16 + r + 8*(lane>=16), col = nt*16 + (lane&15)
  int hi = lane >> 4, nlo = lane & 15;
#define STORE_T(ACC, MT, T)                                         \
  do {                                                              \
    int n = (ngrp * 4 + (T)) * 16 + nlo;                            \
    if (n < ncols) {                                                \
      float* cp = C + (size_t)((MT) * 16 + 8 * hi) * ldc + n;       \
      _Pragma("unroll")                                             \
      for (int r = 0; r < 8; ++r) cp[(size_t)r * ldc] = ACC[r];     \
    }                                                               \
  } while (0)
  STORE_T(c00, mt0, 0); STORE_T(c01, mt0, 1); STORE_T(c02, mt0, 2); STORE_T(c03, mt0, 3);
  STORE_T(c10, mt1, 0); STORE_T(c11, mt1, 1); STORE_T(c12, mt1, 2); STORE_T(c13, mt1, 3);
#undef STORE_T
}

// =====================================================================
// 3) depthwise causal conv over L (kernel 4) + SiLU on the xBC slice
// =====================================================================
__global__ void conv_silu(const float* __restrict__ zx, const float* __restrict__ cw,
                          const float* __restrict__ cb, float* __restrict__ xbc) {
  int idx = blockIdx.x * blockDim.x + threadIdx.x;   // NSEQ*LL*CONV_DIM
  int c = idx & 511;
  int l = (idx >> 9) & 1023;
  int s = idx >> 19;
  const float* base = zx + (size_t)s * LL * LDZ + (size_t)(D_INNER + c);
  float acc = cb[c];
#pragma unroll
  for (int k = 0; k < D_CONV; ++k) {
    int ls = l - (D_CONV - 1) + k;
    if (ls >= 0) acc += base[(size_t)ls * LDZ] * cw[c * D_CONV + k];
  }
  xbc[(size_t)(s * LL + l) * CONV_DIM + c] = siluf(acc);
}

// dt = softplus(dt_raw + dt_bias) ; dA = exp(dt * -exp(A_log))  (padded to 8/head)
__global__ void dt_kernel(const float* __restrict__ zx, const float* __restrict__ dt_bias,
                          const float* __restrict__ A_log,
                          float* __restrict__ dtv, float* __restrict__ dab) {
  int idx = blockIdx.x * blockDim.x + threadIdx.x;   // MROWS*8
  int h = idx & 7;
  int row = idx >> 3;
  float dt = 0.f, da = 0.f;
  if (h < NHEADS) {
    float v = zx[(size_t)row * LDZ + (D_INNER + CONV_DIM) + h] + dt_bias[h];
    dt = (v > 20.f) ? v : log1pf(expf(v));
    da = expf(dt * -expf(A_log[h]));
  }
  dtv[idx] = dt;
  dab[idx] = da;
}

// =====================================================================
// 4) selective scan: 1 block per (seq, head); 64x64 state in registers
//    (16 elems/thread), B/C/x staged in LDS each step, partial-y LDS reduce
// =====================================================================
__global__ void __launch_bounds__(256)
scan_kernel(const float* __restrict__ xbc, const float* __restrict__ dtv,
            const float* __restrict__ dab, const float* __restrict__ D_skip,
            float* __restrict__ y) {
  int s  = blockIdx.x / NHEADS;
  int hh = blockIdx.x % NHEADS;
  int t  = threadIdx.x;
  int p  = t >> 2;                  // headdim index 0..63
  int nq = t & 3;                   // owns states n in [nq*16, nq*16+16)
  __shared__ float Bsh[64], Csh[64], xsh[64], partial[256];
  float hst[16];
#pragma unroll
  for (int i = 0; i < 16; ++i) hst[i] = 0.f;
  float dskip = D_skip[hh];

  for (int l = 0; l < LL; ++l) {
    size_t row = (size_t)s * LL + l;
    const float* xr = xbc + row * CONV_DIM;
    if (t < 64)        Bsh[t]        = xr[D_INNER + t];
    else if (t < 128)  Csh[t - 64]   = xr[D_INNER + D_STATE + (t - 64)];
    else if (t < 192)  xsh[t - 128]  = xr[hh * HEADDIM + (t - 128)];
    __syncthreads();
    float dtl = dtv[row * 8 + hh];
    float dal = dab[row * 8 + hh];
    float dtx = dtl * xsh[p];
    float acc = 0.f;
#pragma unroll
    for (int i = 0; i < 16; ++i) {
      int n = nq * 16 + i;
      float hv = hst[i] * dal + dtx * Bsh[n];
      hst[i] = hv;
      acc += hv * Csh[n];
    }
    partial[t] = acc;
    __syncthreads();
    if (nq == 0) {
      float yv = partial[t] + partial[t + 1] + partial[t + 2] + partial[t + 3]
               + dskip * xsh[p];
      y[row * D_INNER + hh * HEADDIM + p] = yv;
    }
    __syncthreads();
  }
}

// =====================================================================
// 5) y *= silu(z) ; RMSNorm(384) * norm_w   (in place on yscan)
// =====================================================================
__global__ void __launch_bounds__(128)
gate_norm(const float* __restrict__ zx, const float* __restrict__ norm_w,
          float* __restrict__ y) {
  int row = blockIdx.x;
  int t = threadIdx.x;                       // 128 threads, 3 chans each
  __shared__ float red[128];
  const float* zr = zx + (size_t)row * LDZ;
  float* yr = y + (size_t)row * D_INNER;
  float g[3]; float ss = 0.f;
#pragma unroll
  for (int j = 0; j < 3; ++j) {
    int c = t + j * 128;
    float v = yr[c] * siluf(zr[c]);
    g[j] = v;
    ss += v * v;
  }
  red[t] = ss;
  __syncthreads();
  for (int off = 64; off > 0; off >>= 1) {
    if (t < off) red[t] += red[t + off];
    __syncthreads();
  }
  float scale = rsqrtf(red[0] / (float)D_INNER + EPS);
#pragma unroll
  for (int j = 0; j < 3; ++j) {
    int c = t + j * 128;
    yr[c] = g[j] * scale * norm_w[c];
  }
}

// =====================================================================
// 6) inverse-permute the 4 directional outputs, average -> d_out
// =====================================================================
__global__ void combine(const float* __restrict__ seq, float* __restrict__ out) {
  int c = threadIdx.x;                       // 192 threads
  int w = blockIdx.x & 31;
  int h = (blockIdx.x >> 5) & 31;
  int b = blockIdx.x >> 10;
  int l0 = h * 32 + w;
  int l1 = h * 32 + (31 - w);
  int l2 = w * 32 + h;
  int l3 = w * 32 + (31 - h);
  float acc = seq[((size_t)((0 * 4 + b) * LL + l0)) * D_MODEL + c]
            + seq[((size_t)((1 * 4 + b) * LL + l1)) * D_MODEL + c]
            + seq[((size_t)((2 * 4 + b) * LL + l2)) * D_MODEL + c]
            + seq[((size_t)((3 * 4 + b) * LL + l3)) * D_MODEL + c];
  out[(size_t)blockIdx.x * D_MODEL + c] = acc * 0.25f;
}

// ---------------- workspace layout (bytes, all 256-aligned) ----------------
#define OFF_ZX   ((size_t)0)                         // MROWS*912 f32
#define OFF_XBC  (OFF_ZX  + (size_t)MROWS*LDZ*4)     // MROWS*512 f32
#define OFF_DTV  (OFF_XBC + (size_t)MROWS*CONV_DIM*4)
#define OFF_DA   (OFF_DTV + (size_t)MROWS*8*4)
#define OFF_Y    (OFF_DA  + (size_t)MROWS*8*4)       // MROWS*384 f32
#define OFF_SEQ  (OFF_Y   + (size_t)MROWS*D_INNER*4) // MROWS*192 f32
#define OFF_AP1  (OFF_SEQ + (size_t)MROWS*D_MODEL*4) // 6*1024*256 dw
#define OFF_BP1  (OFF_AP1 + (size_t)6*1024*256*4)    // 6*60*256 dw (padded to 60 tiles)
#define OFF_AP2  (OFF_BP1 + (size_t)6*60*256*4)      // 12*1024*256 dw
#define OFF_BP2  (OFF_AP2 + (size_t)12*1024*256*4)   // 12*12*256 dw

extern "C" void kernel_launch(void* const* d_in, const int* in_sizes, int n_in,
                              void* d_out, int out_size, void* d_ws, size_t ws_size,
                              hipStream_t stream) {
  const float* x       = (const float*)d_in[0];
  const float* W_in    = (const float*)d_in[1];
  const float* conv_w  = (const float*)d_in[2];
  const float* conv_b  = (const float*)d_in[3];
  const float* dt_bias = (const float*)d_in[4];
  const float* A_log   = (const float*)d_in[5];
  const float* D_skip  = (const float*)d_in[6];
  const float* norm_w  = (const float*)d_in[7];
  const float* W_out   = (const float*)d_in[8];
  float* out = (float*)d_out;

  char* ws = (char*)d_ws;
  float*    zx    = (float*)(ws + OFF_ZX);
  float*    xbc   = (float*)(ws + OFF_XBC);
  float*    dtv   = (float*)(ws + OFF_DTV);
  float*    dab   = (float*)(ws + OFF_DA);
  float*    ybuf  = (float*)(ws + OFF_Y);
  float*    seq   = (float*)(ws + OFF_SEQ);
  uint32_t* ap1   = (uint32_t*)(ws + OFF_AP1);
  uint32_t* bp1   = (uint32_t*)(ws + OFF_BP1);
  uint32_t* ap2   = (uint32_t*)(ws + OFF_AP2);
  uint32_t* bp2   = (uint32_t*)(ws + OFF_BP2);

  // 1) gather 4 scans -> packed bf16 A (K=192 -> 6 k-blocks)
  gather_pack_A1<<<6 * 1024 * 256 / 256, 256, 0, stream>>>(x, ap1);
  // pack W_in [192][902] -> 60 padded n-tiles ; W_out [384][192] -> 12 n-tiles
  pack_B<<<(6 * 60 * 256) / 256, 256, 0, stream>>>(W_in, bp1, 192, 902, 60);
  pack_B<<<(12 * 12 * 256) / 256, 256, 0, stream>>>(W_out, bp2, 384, 192, 12);
  // 2) in_proj GEMM: 16384 x 902 x 192 (bf16 WMMA, f32 accum)
  //    waves = (1024/2)*15 = 7680 -> 960 blocks of 8 waves (exact)
  gemm_bf16_wmma<<<960, 256, 0, stream>>>(ap1, bp1, zx, 1024, 60, 15, 6, LDZ, D_IN_PROJ);
  // 3) conv + silu ; dt/dA
  conv_silu<<<NSEQ * LL * CONV_DIM / 256, 256, 0, stream>>>(zx, conv_w, conv_b, xbc);
  dt_kernel<<<MROWS * 8 / 256, 256, 0, stream>>>(zx, dt_bias, A_log, dtv, dab);
  // 4) selective scan (96 independent (seq,head) streams)
  scan_kernel<<<NSEQ * NHEADS, 256, 0, stream>>>(xbc, dtv, dab, D_skip, ybuf);
  // 5) gate + RMSNorm (in place)
  gate_norm<<<MROWS, 128, 0, stream>>>(zx, norm_w, ybuf);
  // 6) out_proj GEMM: 16384 x 192 x 384 ; waves = 512*3 = 1536 -> 192 blocks
  pack_A<<<12 * 1024 * 256 / 256, 256, 0, stream>>>(ybuf, ap2, D_INNER);
  gemm_bf16_wmma<<<192, 256, 0, stream>>>(ap2, bp2, seq, 1024, 12, 3, 12, D_MODEL, D_MODEL);
  // 7) inverse permute + average
  combine<<<BB * HH * WW, D_MODEL, 0, stream>>>(seq, out);
}
